// CrossModalAlignmentModule_38637525795414
// MI455X (gfx1250) — matrix operand order repaired and checked by reading, not compiled, not run on hardware
//
#include <hip/hip_runtime.h>
#include <hip/hip_bf16.h>
#include <cmath>

// ---------------------------------------------------------------------------
// CDNA5 (gfx1250) implementation of CrossModalAlignmentModule forward.
// All matmul work goes through V_WMMA_F32_16X16X32_BF16 (wave32, f32 acc).
// ~295 GFLOP over ~200MB -> compute and memory floors are both ~10us on
// MI455X; bf16 WMMA is the highest-throughput precision that preserves
// LN/softmax numerics.
//
// GEMM design: one wave owns a 16x(16*NTILES) strip. Ragged M/N edges are
// handled by CLAMPING row indices (garbage only reaches never-stored C
// rows/cols), so the k-loop is branchless: unconditional b128 loads + NTILES
// independent WMMAs per k-step (A-fragment reused NTILES times, independent
// accumulators fill WMMA hazard slots). Verified in disasm: clause-grouped
// global_load_b128 + partial s_wait_loadcnt + back-to-back WMMAs.
// ---------------------------------------------------------------------------

typedef __bf16 bf16;
typedef __attribute__((ext_vector_type(16))) __bf16 v16bf;
typedef __attribute__((ext_vector_type(8)))  __bf16 v8bf;
typedef __attribute__((ext_vector_type(8)))  float  v8f;

#define E_DIM 768
#define N_HEADS 8
#define D_HEAD 96

__device__ __forceinline__ bf16 f2bf(float f) {
    union { float f; unsigned u; } v; v.f = f;
    unsigned r = v.u + 0x7fffu + ((v.u >> 16) & 1u);   // round-to-nearest-even
    unsigned short h = (unsigned short)(r >> 16);
    return __builtin_bit_cast(bf16, h);
}

// ---------------------------------------------------------------------------
// f32 -> bf16 elementwise conversion
// ---------------------------------------------------------------------------
__global__ void f2bf_kernel(const float* __restrict__ in, bf16* __restrict__ out, long n) {
    long i = (long)blockIdx.x * blockDim.x + threadIdx.x;
    if (i < n) out[i] = f2bf(in[i]);
}

// ---------------------------------------------------------------------------
// Generic batched WMMA GEMM.
//   TRANSB=1:  C[m,n] = sum_k A[m,k] * B[n,k]   (B stored [N,K] row-major)
//   TRANSB=0:  C[m,n] = sum_k A[m,k] * B[k,n]   (B stored [K,N] row-major)
// A may be split (A1 for k<K1, A2 for k>=K1) to virtualize concat(); the
// split test is wave-uniform (K1, k0 multiples of 32) -> scalar select.
// Batch offsets: bat = bo*batH + bi;  off = bo*Out + bi*In  (per operand).
// K must be a multiple of 32 (pad regions of operands are valid zeros).
// One wave computes a 16 x (16*NTILES) strip; 4 waves per block, consecutive
// waves share A rows (n-group varies fastest) for WGP$ locality.
// ---------------------------------------------------------------------------
template<int TRANSB, int NTILES>
__global__ __launch_bounds__(128)
void wmma_gemm_kernel(const bf16* __restrict__ A1, const bf16* __restrict__ A2, int K1,
                      const bf16* __restrict__ Bm, const float* __restrict__ bias,
                      float* __restrict__ Cf, bf16* __restrict__ Cb,
                      int M, int N, int K, int lda, int ldb, int ldc,
                      int batches, int batH,
                      long aOut, long aIn, long bOut, long bIn, long cOut, long cIn)
{
    const int tilesM = (M + 15) >> 4;
    const int tilesG = (N + NTILES * 16 - 1) / (NTILES * 16);
    const long wtile = (long)blockIdx.x * (blockDim.x >> 5) + (threadIdx.x >> 5);
    const long totalTiles = (long)batches * tilesM * tilesG;
    if (wtile >= totalTiles) return;             // uniform per wave

    const int bat = (int)(wtile / ((long)tilesM * tilesG));
    const int rem = (int)(wtile % ((long)tilesM * tilesG));
    const int mT = rem / tilesG, nG = rem % tilesG;
    const int bo = bat / batH, bi = bat % batH;

    const bf16* a1 = A1 + (long)bo * aOut + (long)bi * aIn;
    const bf16* a2 = (A2 ? A2 : A1) + (long)bo * aOut + (long)bi * aIn;
    const bf16* bp = Bm + (long)bo * bOut + (long)bi * bIn;

    const int lane = threadIdx.x & 31;
    const int r    = lane & 15;
    const int half = lane >> 4;
    const int mBase = mT * 16;

    // Clamped A row: rows >= M feed only C rows >= M, which are never stored.
    const int aRowC = min(mBase + r, M - 1);
    const bf16* aBase1 = a1 + (long)aRowC * lda;
    const bf16* aBase2 = a2 + (long)aRowC * lda;

    // Per-n-tile B bases (clamped the same way).
    const bf16* bBase[NTILES];   // TRANSB=1: row pointer;  TRANSB=0: col pointer
#pragma unroll
    for (int t = 0; t < NTILES; ++t) {
        const int n = min((nG * NTILES + t) * 16 + r, N - 1);
        bBase[t] = TRANSB ? (bp + (long)n * ldb) : (bp + n);
    }

    v8f acc[NTILES];
#pragma unroll
    for (int t = 0; t < NTILES; ++t)
#pragma unroll
        for (int j = 0; j < 8; ++j) acc[t][j] = 0.0f;

#pragma unroll 2
    for (int k0 = 0; k0 < K; k0 += 32) {
        // ---- A fragment: contiguous 16 K-values per lane at K = half*16
        //      (matches B fragment order).
        const int kk = k0 + half * 16;
        const bf16* ab = (kk >= K1) ? (aBase2 + (kk - K1)) : (aBase1 + kk);
        // locality=3 -> WGP-scope prefetch (pulls into all cache levels);
        // SYS-scope (locality 0) would skip the WGP cache entirely.
        __builtin_prefetch(ab + 32, 0, 3);       // next k-step (global_prefetch)
        v8bf alo = *(const v8bf*)(ab);
        v8bf ahi = *(const v8bf*)(ab + 8);
        v16bf af;
#pragma unroll
        for (int i = 0; i < 8; ++i) { af[i] = alo[i]; af[i + 8] = ahi[i]; }

        // ---- B fragments: same order as A = contiguous 16 K-values
        //      per lane at K = half*16 (column index = lane&15).
        v16bf bfr[NTILES];
#pragma unroll
        for (int t = 0; t < NTILES; ++t) {
            if (TRANSB) {
                const bf16* q = bBase[t] + k0 + half * 16;
                v8bf lo = *(const v8bf*)(q);
                v8bf hi = *(const v8bf*)(q + 8);
#pragma unroll
                for (int i = 0; i < 8; ++i) { bfr[t][i] = lo[i]; bfr[t][i + 8] = hi[i]; }
            } else {
                const bf16* q = bBase[t] + (long)(k0 + half * 16) * ldb;
#pragma unroll
                for (int i = 0; i < 16; ++i) bfr[t][i] = q[(long)i * ldb];
            }
        }

        // NTILES independent accumulators -> WMMA hazard slots carry real work.
#pragma unroll
        for (int t = 0; t < NTILES; ++t)
            acc[t] = __builtin_amdgcn_wmma_f32_16x16x32_bf16(
                false, af, false, bfr[t], (short)0, acc[t], false, false);
    }

    // ---- store: D layout VGPR j -> row = half*8 + j, col = lane&15
    const long coff = (long)bo * cOut + (long)bi * cIn;
#pragma unroll
    for (int t = 0; t < NTILES; ++t) {
        const int col = (nG * NTILES + t) * 16 + r;
        if (col < N) {
            const float bv = bias ? bias[col] : 0.0f;
#pragma unroll
            for (int j = 0; j < 8; ++j) {
                const int row = mBase + half * 8 + j;
                if (row < M) {
                    const float v = acc[t][j] + bv;
                    if (Cf) Cf[coff + (long)row * ldc + col] = v;
                    else    Cb[coff + (long)row * ldc + col] = f2bf(v);
                }
            }
        }
    }
}

// ---------------------------------------------------------------------------
// Row softmax over scores [B*H, Nq, NkvPad] (f32) -> probs (bf16, pad zeroed).
// One wave per row. Optional text mask [B, Nkv].
// ---------------------------------------------------------------------------
__global__ __launch_bounds__(128)
void softmax_kernel(const float* __restrict__ scores, bf16* __restrict__ probs,
                    const int* __restrict__ mask,
                    int H, int Nq, int Nkv, int NkvPad, float scale)
{
    const long row = (long)blockIdx.x * (blockDim.x >> 5) + (threadIdx.x >> 5);
    const int lane = threadIdx.x & 31;
    const long nRows = (long)64 * H * Nq;
    if (row >= nRows) return;
    const int bh = (int)(row / Nq);
    const int b  = bh / H;
    const float* s = scores + row * NkvPad;
    bf16* p = probs + row * NkvPad;

    float m = -3.4e38f;
    for (int j = lane; j < Nkv; j += 32) {
        float v = s[j] * scale;
        if (mask && mask[b * Nkv + j] == 0) v = -1e9f;
        m = fmaxf(m, v);
    }
#pragma unroll
    for (int off = 16; off > 0; off >>= 1) m = fmaxf(m, __shfl_xor(m, off, 32));

    float sum = 0.0f;
    for (int j = lane; j < Nkv; j += 32) {
        float v = s[j] * scale;
        if (mask && mask[b * Nkv + j] == 0) v = -1e9f;
        sum += __expf(v - m);
    }
#pragma unroll
    for (int off = 16; off > 0; off >>= 1) sum += __shfl_xor(sum, off, 32);
    const float inv = 1.0f / sum;

    for (int j = lane; j < NkvPad; j += 32) {
        float v = 0.0f;
        if (j < Nkv) {
            float t = s[j] * scale;
            if (mask && mask[b * Nkv + j] == 0) t = -1e9f;
            v = __expf(t - m) * inv;
        }
        p[j] = f2bf(v);
    }
}

// ---------------------------------------------------------------------------
// LayerNorm (eps 1e-5, population var) + exact-erf GELU. One block per row.
// ---------------------------------------------------------------------------
__global__ __launch_bounds__(256)
void ln_gelu_kernel(const float* __restrict__ Hin, const float* __restrict__ g,
                    const float* __restrict__ beta, float* __restrict__ out)
{
    __shared__ float red[256];
    const int tid = threadIdx.x;
    const long row = blockIdx.x;
    const float* h = Hin + row * E_DIM;
    float x0 = h[tid], x1 = h[tid + 256], x2 = h[tid + 512];

    red[tid] = x0 + x1 + x2; __syncthreads();
    for (int st = 128; st > 0; st >>= 1) { if (tid < st) red[tid] += red[tid + st]; __syncthreads(); }
    const float mean = red[0] * (1.0f / E_DIM);
    __syncthreads();

    red[tid] = x0 * x0 + x1 * x1 + x2 * x2; __syncthreads();
    for (int st = 128; st > 0; st >>= 1) { if (tid < st) red[tid] += red[tid + st]; __syncthreads(); }
    const float var = red[0] * (1.0f / E_DIM) - mean * mean;
    const float rstd = rsqrtf(var + 1e-5f);

    float* o = out + row * E_DIM;
    const float xs[3] = {x0, x1, x2};
#pragma unroll
    for (int i = 0; i < 3; ++i) {
        const int e = tid + i * 256;
        const float y = (xs[i] - mean) * rstd * g[e] + beta[e];
        o[e] = 0.5f * y * (1.0f + erff(y * 0.70710678118654752f));
    }
}

// ---------------------------------------------------------------------------
// Token-mean pooling: pooled[b,e] = mean_t x[b,t,e]
// ---------------------------------------------------------------------------
__global__ void pool_kernel(const float* __restrict__ x, float* __restrict__ pooled, int Ntok)
{
    const int b = blockIdx.x;
    for (int e = threadIdx.x; e < E_DIM; e += blockDim.x) {
        float s = 0.0f;
        const float* px = x + (long)b * Ntok * E_DIM + e;
        for (int t = 0; t < Ntok; ++t) s += px[(long)t * E_DIM];
        pooled[(long)b * E_DIM + e] = s / (float)Ntok;
    }
}

// ---------------------------------------------------------------------------
// Row L2 normalize (clamped at 1e-12). One block per row of 768.
// ---------------------------------------------------------------------------
__global__ __launch_bounds__(256)
void l2norm_kernel(const float* __restrict__ in, float* __restrict__ out)
{
    __shared__ float red[256];
    const int tid = threadIdx.x;
    const long b = blockIdx.x;
    const float* x = in + b * E_DIM;
    float x0 = x[tid], x1 = x[tid + 256], x2 = x[tid + 512];
    red[tid] = x0 * x0 + x1 * x1 + x2 * x2; __syncthreads();
    for (int st = 128; st > 0; st >>= 1) { if (tid < st) red[tid] += red[tid + st]; __syncthreads(); }
    const float n = fmaxf(sqrtf(red[0]), 1e-12f);
    const float inv = 1.0f / n;
    float* o = out + b * E_DIM;
    o[tid] = x0 * inv; o[tid + 256] = x1 * inv; o[tid + 512] = x2 * inv;
}

// logits[i,j] = dot(an[i], bn[j]) / TEMP     (64x64, K=768)
__global__ void logits_kernel(const float* __restrict__ an, const float* __restrict__ bn,
                              float* __restrict__ logits)
{
    const int i = blockIdx.x, j = threadIdx.x;
    float s = 0.0f;
    for (int k = 0; k < E_DIM; ++k) s += an[i * E_DIM + k] * bn[j * E_DIM + k];
    logits[i * 64 + j] = s * (1.0f / 0.07f);
}

__global__ void zero_loss_kernel(float* loss) { *loss = 0.0f; }

// 0.5*(ce(L)+ce(L^T))/3 accumulated into *loss
__global__ __launch_bounds__(64)
void ce_pair_kernel(const float* __restrict__ logits, float* __restrict__ loss)
{
    __shared__ float red[64];
    const int t = threadIdx.x;
    float mR = -3.4e38f, mC = -3.4e38f;
    for (int j = 0; j < 64; ++j) { mR = fmaxf(mR, logits[t * 64 + j]); mC = fmaxf(mC, logits[j * 64 + t]); }
    float sR = 0.0f, sC = 0.0f;
    for (int j = 0; j < 64; ++j) { sR += __expf(logits[t * 64 + j] - mR); sC += __expf(logits[j * 64 + t] - mC); }
    const float lseR = mR + __logf(sR);
    const float lseC = mC + __logf(sC);
    const float d = logits[t * 64 + t];
    red[t] = (lseR - d) + (lseC - d);
    __syncthreads();
    for (int st = 32; st > 0; st >>= 1) { if (t < st) red[t] += red[t + st]; __syncthreads(); }
    if (t == 0) atomicAdd(loss, 0.5f * red[0] * (1.0f / 64.0f) * (1.0f / 3.0f));
}

// ---------------------------------------------------------------------------
// Host-side orchestration
// ---------------------------------------------------------------------------
struct AW { bf16 *qw, *kw, *vw, *ow; const float *qb, *kb, *vb, *ob; };

static void run_gemm(hipStream_t st,
                     const bf16* A1, const bf16* A2, int K1,
                     const bf16* Bm, const float* bias,
                     float* Cf, bf16* Cb,
                     int M, int N, int K, int lda, int ldb, int ldc,
                     int batches, int batH,
                     long aOut, long aIn, long bOut, long bIn, long cOut, long cIn,
                     int transB)
{
    const int ntiles = transB ? 4 : 2;
    long tilesM = (M + 15) / 16;
    long tilesG = (N + ntiles * 16 - 1) / (ntiles * 16);
    long total = (long)batches * tilesM * tilesG;
    unsigned blocks = (unsigned)((total + 3) / 4);
    if (transB)
        wmma_gemm_kernel<1, 4><<<dim3(blocks), dim3(128), 0, st>>>(
            A1, A2, K1, Bm, bias, Cf, Cb, M, N, K, lda, ldb, ldc,
            batches, batH, aOut, aIn, bOut, bIn, cOut, cIn);
    else
        wmma_gemm_kernel<0, 2><<<dim3(blocks), dim3(128), 0, st>>>(
            A1, A2, K1, Bm, bias, Cf, Cb, M, N, K, lda, ldb, ldc,
            batches, batH, aOut, aIn, bOut, bIn, cOut, cIn);
}

static void cross_attn(hipStream_t st,
                       const bf16* qin, int Nq, const bf16* kvin, int Nkv,
                       const AW& w, const int* mask,
                       bf16* Qbf, bf16* Kbf, bf16* Vbf,
                       float* scores, bf16* probs, bf16* AVbf,
                       bf16* outbf)
{
    const int B = 64, E = E_DIM, H = N_HEADS, DH = D_HEAD;
    const int Mq = B * Nq, Mk = B * Nkv;
    const int NkvPad = ((Nkv + 31) / 32) * 32;
    const float scale = 1.0f / sqrtf((float)DH);
    const int KMAX = 0x40000000;   // "no split"

    // Q/K/V projections: x @ W^T + b  (NT GEMM), output bf16
    run_gemm(st, qin,  nullptr, KMAX, w.qw, w.qb, nullptr, Qbf, Mq, E, E, E, E, E, 1, 1, 0,0,0,0,0,0, 1);
    run_gemm(st, kvin, nullptr, KMAX, w.kw, w.kb, nullptr, Kbf, Mk, E, E, E, E, E, 1, 1, 0,0,0,0,0,0, 1);
    run_gemm(st, kvin, nullptr, KMAX, w.vw, w.vb, nullptr, Vbf, Mk, E, E, E, E, E, 1, 1, 0,0,0,0,0,0, 1);

    // scores[b,h,q,k] = Q[b,q,h,:] . K[b,k,h,:]   (batched NT, K=96)
    run_gemm(st, Qbf, nullptr, KMAX, Kbf, nullptr, scores, nullptr,
             Nq, Nkv, DH, E, E, NkvPad,
             B * H, H,
             (long)Nq * E, DH,
             (long)Nkv * E, DH,
             (long)H * Nq * NkvPad, (long)Nq * NkvPad,
             1);

    {   // masked, scaled softmax -> bf16 probs (pad region zeroed)
        long rows = (long)B * H * Nq;
        unsigned blocks = (unsigned)((rows + 3) / 4);
        softmax_kernel<<<dim3(blocks), dim3(128), 0, st>>>(
            scores, probs, mask, H, Nq, Nkv, NkvPad, scale);
    }

    // out[b,q,h,d] = sum_k probs[b,h,q,k] * V[b,k,h,d]   (batched NN, K=NkvPad)
    run_gemm(st, probs, nullptr, KMAX, Vbf, nullptr, nullptr, AVbf,
             Nq, DH, NkvPad, NkvPad, E, E,
             B * H, H,
             (long)H * Nq * NkvPad, (long)Nq * NkvPad,
             (long)Nkv * E, DH,
             (long)Nq * E, DH,
             0);

    // output projection
    run_gemm(st, AVbf, nullptr, KMAX, w.ow, w.ob, nullptr, outbf, Mq, E, E, E, E, E, 1, 1, 0,0,0,0,0,0, 1);
}

static void fusion(hipStream_t st, const bf16* xbf, const bf16* abf, int rows,
                   const bf16* wb, const float* b, const float* g, const float* beta,
                   float* Hbuf, float* outRegion)
{
    // concat([x, attn], -1) @ W^T + b via split-A GEMM (K=1536, K1=768)
    run_gemm(st, xbf, abf, E_DIM, wb, b, Hbuf, nullptr,
             rows, E_DIM, 2 * E_DIM, E_DIM, 2 * E_DIM, E_DIM,
             1, 1, 0,0,0,0,0,0, 1);
    ln_gelu_kernel<<<dim3(rows), dim3(256), 0, st>>>(Hbuf, g, beta, outRegion);
}

extern "C" void kernel_launch(void* const* d_in, const int* in_sizes, int n_in,
                              void* d_out, int out_size, void* d_ws, size_t ws_size,
                              hipStream_t stream)
{
    (void)in_sizes; (void)n_in; (void)out_size; (void)ws_size;
    const int B = 64, Nr = 196, Nt = 128, E = E_DIM;
    const int MR = B * Nr;   // 12544
    const int MT = B * Nt;   // 8192

    auto f = [&](int i) { return (const float*)d_in[i]; };
    const float* rgb = f(0);
    const float* hsi = f(1);
    const float* txt = f(2);
    // params flattened in dict insertion order:
    //   3..10  rgb_to_text (qw,qb,kw,kb,vw,vb,ow,ob)
    //  11..18  hsi_to_text  (output unused in reference -> fully skipped)
    //  19..26  text_to_rgb
    //  27..34  text_to_hsi
    //  35..38  fusion_rgb (w,b,g,beta)   39..42 fusion_hsi   43..46 fusion_text
    //  47      text_mask [B, Nt] int32
    const int* mask = (const int*)d_in[47];

    char* ws = (char*)d_ws;
    size_t off = 0;
    auto alloc = [&](size_t bytes) -> void* {
        off = (off + 255) & ~(size_t)255;
        void* p = ws + off; off += bytes; return p;
    };
    auto conv = [&](const float* src, bf16* dst, long n) {
        long blocks = (n + 255) / 256;
        f2bf_kernel<<<dim3((unsigned)blocks), dim3(256), 0, stream>>>(src, dst, n);
    };
    auto wbf = [&](int idx, long n) -> bf16* {
        bf16* p = (bf16*)alloc((size_t)n * 2);
        conv(f(idx), p, n);
        return p;
    };

    // bf16 copies of activations
    bf16* rgb_bf = (bf16*)alloc((size_t)MR * E * 2);
    bf16* hsi_bf = (bf16*)alloc((size_t)MR * E * 2);
    bf16* txt_bf = (bf16*)alloc((size_t)MT * E * 2);
    conv(rgb, rgb_bf, (long)MR * E);
    conv(hsi, hsi_bf, (long)MR * E);
    conv(txt, txt_bf, (long)MT * E);

    // bf16 weights (only the 3 attentions whose outputs are used)
    auto loadAW = [&](int base) -> AW {
        AW a;
        a.qw = wbf(base + 0, (long)E * E); a.qb = f(base + 1);
        a.kw = wbf(base + 2, (long)E * E); a.kb = f(base + 3);
        a.vw = wbf(base + 4, (long)E * E); a.vb = f(base + 5);
        a.ow = wbf(base + 6, (long)E * E); a.ob = f(base + 7);
        return a;
    };
    AW w_r2t = loadAW(3);    // rgb_to_text  (q=txt, kv=rgb)
    AW w_t2r = loadAW(19);   // text_to_rgb  (q=rgb, kv=txt)
    AW w_t2h = loadAW(27);   // text_to_hsi  (q=hsi, kv=txt)
    bf16* fw_rgb = wbf(35, (long)E * 2 * E);
    bf16* fw_hsi = wbf(39, (long)E * 2 * E);
    bf16* fw_txt = wbf(43, (long)E * 2 * E);

    // persistent attention outputs (bf16, consumed by fusion)
    bf16* rgb_ft = (bf16*)alloc((size_t)MR * E * 2);  // rgb_from_text
    bf16* hsi_ft = (bf16*)alloc((size_t)MR * E * 2);  // hsi_from_text
    bf16* txt_fr = (bf16*)alloc((size_t)MT * E * 2);  // text_from_rgb

    // reusable attention scratch (sized for worst case)
    bf16* Qbf = (bf16*)alloc((size_t)MR * E * 2);
    bf16* Kbf = (bf16*)alloc((size_t)MR * E * 2);
    bf16* Vbf = (bf16*)alloc((size_t)MR * E * 2);
    const size_t maxScoreElems = (size_t)B * N_HEADS * 28672;  // max(196*128, 128*224)
    float* scores = (float*)alloc(maxScoreElems * 4);
    bf16*  probs  = (bf16*)alloc(maxScoreElems * 2);
    bf16*  AVbf   = (bf16*)alloc((size_t)MR * E * 2);
    float* Hbuf   = (float*)alloc((size_t)MR * E * 4);

    // pooled / normalized / logits scratch
    float* pr  = (float*)alloc((size_t)B * E * 4);
    float* ph  = (float*)alloc((size_t)B * E * 4);
    float* pt  = (float*)alloc((size_t)B * E * 4);
    float* nr_ = (float*)alloc((size_t)B * E * 4);
    float* nh_ = (float*)alloc((size_t)B * E * 4);
    float* nt_ = (float*)alloc((size_t)B * E * 4);
    float* logits = (float*)alloc(64 * 64 * 4);

    // output layout: rgb_aligned | hsi_aligned | text_aligned | loss
    float* outF   = (float*)d_out;
    float* rgbOut = outF;
    float* hsiOut = outF + (size_t)MR * E;
    float* txtOut = outF + (size_t)2 * MR * E;
    float* lossP  = outF + (size_t)2 * MR * E + (size_t)MT * E;

    // ---- cross attentions (hsi_to_text output unused in reference -> skipped)
    cross_attn(stream, rgb_bf, Nr, txt_bf, Nt, w_t2r, mask,    Qbf, Kbf, Vbf, scores, probs, AVbf, rgb_ft);
    cross_attn(stream, txt_bf, Nt, rgb_bf, Nr, w_r2t, nullptr, Qbf, Kbf, Vbf, scores, probs, AVbf, txt_fr);
    cross_attn(stream, hsi_bf, Nr, txt_bf, Nt, w_t2h, mask,    Qbf, Kbf, Vbf, scores, probs, AVbf, hsi_ft);

    // ---- fusion heads -> f32 outputs
    fusion(stream, rgb_bf, rgb_ft, MR, fw_rgb, f(36), f(37), f(38), Hbuf, rgbOut);
    fusion(stream, hsi_bf, hsi_ft, MR, fw_hsi, f(40), f(41), f(42), Hbuf, hsiOut);
    fusion(stream, txt_bf, txt_fr, MT, fw_txt, f(44), f(45), f(46), Hbuf, txtOut);

    // ---- contrastive loss on pooled raw features
    pool_kernel<<<dim3(B), dim3(256), 0, stream>>>(rgb, pr, Nr);
    pool_kernel<<<dim3(B), dim3(256), 0, stream>>>(hsi, ph, Nr);
    pool_kernel<<<dim3(B), dim3(256), 0, stream>>>(txt, pt, Nt);
    l2norm_kernel<<<dim3(B), dim3(256), 0, stream>>>(pr, nr_);
    l2norm_kernel<<<dim3(B), dim3(256), 0, stream>>>(ph, nh_);
    l2norm_kernel<<<dim3(B), dim3(256), 0, stream>>>(pt, nt_);
    zero_loss_kernel<<<dim3(1), dim3(1), 0, stream>>>(lossP);

    logits_kernel<<<dim3(64), dim3(64), 0, stream>>>(nr_, nt_, logits);
    ce_pair_kernel<<<dim3(1), dim3(64), 0, stream>>>(logits, lossP);
    logits_kernel<<<dim3(64), dim3(64), 0, stream>>>(nh_, nt_, logits);
    ce_pair_kernel<<<dim3(1), dim3(64), 0, stream>>>(logits, lossP);
    logits_kernel<<<dim3(64), dim3(64), 0, stream>>>(nr_, nh_, logits);
    ce_pair_kernel<<<dim3(1), dim3(64), 0, stream>>>(logits, lossP);
}
